// KoLeoLossDistributed_44573170598815
// MI455X (gfx1250) — matrix operand order repaired and checked by reading, not compiled, not run on hardware
//
#include <hip/hip_runtime.h>
#include <math.h>

#define B_ROWS 8192
#define DIMS   1024
#define NCHUNKS 16
#define COLS_PER_CHUNK (B_ROWS / NCHUNKS)   // 512
#define WAVES_PER_WG 8
#define ROWS_PER_WAVE 32
#define EPSF 1e-8f
#define GATE_THRESHOLD_F 0.5f
#define GATE_ALPHA_F 0.1f

typedef __attribute__((ext_vector_type(16))) __bf16 v16bf;
typedef __attribute__((ext_vector_type(8)))  float  v8f;

union V16 {
  v16bf v;
  uint4 q[2];
};

// round-to-nearest-even f32 -> bf16 (inputs are finite normalized values)
__device__ __forceinline__ unsigned short f2bf(float f) {
  unsigned int u = __float_as_uint(f);
  u += 0x7fffu + ((u >> 16) & 1u);
  return (unsigned short)(u >> 16);
}

// total-order "better" predicate: larger value wins; equal value -> lower index
__device__ __forceinline__ bool better(float v, int idx, float bv, int bidx) {
  return (v > bv) || ((v == bv) && (idx < bidx));
}

// ---------------------------------------------------------------------------
// Kernel 1: row L2 norm, write invnorm (f32) and xn (bf16). One block per row,
// one float4 per thread.
// ---------------------------------------------------------------------------
__global__ void __launch_bounds__(256)
normalize_bf16(const float* __restrict__ x, unsigned short* __restrict__ xnbf,
               float* __restrict__ invn) {
  const int row = blockIdx.x;
  const float4* xr4 = reinterpret_cast<const float4*>(x + (size_t)row * DIMS);
  const float4 v4 = xr4[threadIdx.x];                 // 256 * 4 = 1024
  float ss = v4.x * v4.x + v4.y * v4.y + v4.z * v4.z + v4.w * v4.w;
  for (int o = 16; o > 0; o >>= 1) ss += __shfl_down(ss, o, 32);
  __shared__ float red[8];
  __shared__ float sInv;
  if ((threadIdx.x & 31) == 0) red[threadIdx.x >> 5] = ss;
  __syncthreads();
  if (threadIdx.x == 0) {
    float t = 0.f;
    for (int w = 0; w < 8; ++w) t += red[w];
    float inv = 1.0f / fmaxf(sqrtf(t), EPSF);         // xn = x / max(||x||, eps)
    sInv = inv;
    invn[row] = inv;
  }
  __syncthreads();
  const float inv = sInv;
  uint2 packed;
  packed.x = (unsigned)f2bf(v4.x * inv) | ((unsigned)f2bf(v4.y * inv) << 16);
  packed.y = (unsigned)f2bf(v4.z * inv) | ((unsigned)f2bf(v4.w * inv) << 16);
  reinterpret_cast<uint2*>(xnbf + (size_t)row * DIMS)[threadIdx.x] = packed;
}

// ---------------------------------------------------------------------------
// Kernel 2: fused bf16-WMMA GEMM (dots = xn @ xn^T) + per-row running top-2.
// Each wave owns a 32-row band and computes 32x64 register blocks (2 A tiles x
// 4 B tiles = 8 WMMAs/K-step, 12 loads/K-step => 1.5 loads/WMMA) across a
// 512-column chunk. The K loop is double-buffered so loads for step kb+32 are
// in flight while the eight WMMAs for step kb execute. Top-2 scan is
// branchless with all 32 lanes active (lane owns rows lane&15 and 16+(lane&15)
// of the band, column-half kh).
// ---------------------------------------------------------------------------
__global__ void __launch_bounds__(256)
koleo_topk_wmma(const unsigned short* __restrict__ xnbf,
                float* __restrict__ pVal, int* __restrict__ pIdx) {
  __shared__ float tile[WAVES_PER_WG][16][17];   // padded: no bank conflicts
  const int wave = threadIdx.x >> 5;
  const int lane = threadIdx.x & 31;
  const int kh   = lane >> 4;     // half selector per ISA A/B striping
  const int lr   = lane & 15;
  const int rowBase  = (blockIdx.x * WAVES_PER_WG + wave) * ROWS_PER_WAVE;
  const int colBase0 = blockIdx.y * COLS_PER_CHUNK;
  const int grow0 = rowBase + lr;        // row owned in band 0 scan
  const int grow1 = rowBase + 16 + lr;   // row owned in band 1 scan

  // A operand: lane holds row (base+lr); elems 0..15 = K kh*16..+15
  // within each 32-wide K step.
  const unsigned short* arow0 = xnbf + (size_t)grow0 * DIMS + kh * 16;
  const unsigned short* arow1 = xnbf + (size_t)grow1 * DIMS + kh * 16;

  float a_b1 = -2.f, a_b2 = -2.f; int a_i1 = 0, a_i2 = 0;  // band 0 top-2
  float c_b1 = -2.f, c_b2 = -2.f; int c_i1 = 0, c_i2 = 0;  // band 1 top-2

  for (int g = 0; g < COLS_PER_CHUNK / 64; ++g) {
    const int cb = colBase0 + g * 64;
    // B operand (column-major xn^T == row-major xn): lane holds 16
    // consecutive K of column (cb + s*16 + lr), starting at kh*16.
    const unsigned short* bb0 = xnbf + (size_t)(cb + 0  + lr) * DIMS + kh * 16;
    const unsigned short* bb1 = xnbf + (size_t)(cb + 16 + lr) * DIMS + kh * 16;
    const unsigned short* bb2 = xnbf + (size_t)(cb + 32 + lr) * DIMS + kh * 16;
    const unsigned short* bb3 = xnbf + (size_t)(cb + 48 + lr) * DIMS + kh * 16;

    v8f acc[8] = {};

    // --- prologue: load K step 0 ---
    V16 A0[2], B0[4];
    A0[0].q[0] = reinterpret_cast<const uint4*>(arow0)[0];
    A0[0].q[1] = reinterpret_cast<const uint4*>(arow0)[1];
    A0[1].q[0] = reinterpret_cast<const uint4*>(arow1)[0];
    A0[1].q[1] = reinterpret_cast<const uint4*>(arow1)[1];
    B0[0].q[0] = reinterpret_cast<const uint4*>(bb0)[0];
    B0[0].q[1] = reinterpret_cast<const uint4*>(bb0)[1];
    B0[1].q[0] = reinterpret_cast<const uint4*>(bb1)[0];
    B0[1].q[1] = reinterpret_cast<const uint4*>(bb1)[1];
    B0[2].q[0] = reinterpret_cast<const uint4*>(bb2)[0];
    B0[2].q[1] = reinterpret_cast<const uint4*>(bb2)[1];
    B0[3].q[0] = reinterpret_cast<const uint4*>(bb3)[0];
    B0[3].q[1] = reinterpret_cast<const uint4*>(bb3)[1];

#pragma unroll 2
    for (int kb = 0; kb < DIMS; kb += 32) {
      // prefetch next K step (wraps to 0 on the last step; harmless in-bounds)
      const int kn = (kb + 32 < DIMS) ? (kb + 32) : 0;
      V16 A1[2], B1[4];
      A1[0].q[0] = reinterpret_cast<const uint4*>(arow0 + kn)[0];
      A1[0].q[1] = reinterpret_cast<const uint4*>(arow0 + kn)[1];
      A1[1].q[0] = reinterpret_cast<const uint4*>(arow1 + kn)[0];
      A1[1].q[1] = reinterpret_cast<const uint4*>(arow1 + kn)[1];
      B1[0].q[0] = reinterpret_cast<const uint4*>(bb0 + kn)[0];
      B1[0].q[1] = reinterpret_cast<const uint4*>(bb0 + kn)[1];
      B1[1].q[0] = reinterpret_cast<const uint4*>(bb1 + kn)[0];
      B1[1].q[1] = reinterpret_cast<const uint4*>(bb1 + kn)[1];
      B1[2].q[0] = reinterpret_cast<const uint4*>(bb2 + kn)[0];
      B1[2].q[1] = reinterpret_cast<const uint4*>(bb2 + kn)[1];
      B1[3].q[0] = reinterpret_cast<const uint4*>(bb3 + kn)[0];
      B1[3].q[1] = reinterpret_cast<const uint4*>(bb3 + kn)[1];

      acc[0] = __builtin_amdgcn_wmma_f32_16x16x32_bf16(
          false, A0[0].v, false, B0[0].v, (short)0, acc[0], false, false);
      acc[1] = __builtin_amdgcn_wmma_f32_16x16x32_bf16(
          false, A0[0].v, false, B0[1].v, (short)0, acc[1], false, false);
      acc[2] = __builtin_amdgcn_wmma_f32_16x16x32_bf16(
          false, A0[0].v, false, B0[2].v, (short)0, acc[2], false, false);
      acc[3] = __builtin_amdgcn_wmma_f32_16x16x32_bf16(
          false, A0[0].v, false, B0[3].v, (short)0, acc[3], false, false);
      acc[4] = __builtin_amdgcn_wmma_f32_16x16x32_bf16(
          false, A0[1].v, false, B0[0].v, (short)0, acc[4], false, false);
      acc[5] = __builtin_amdgcn_wmma_f32_16x16x32_bf16(
          false, A0[1].v, false, B0[1].v, (short)0, acc[5], false, false);
      acc[6] = __builtin_amdgcn_wmma_f32_16x16x32_bf16(
          false, A0[1].v, false, B0[2].v, (short)0, acc[6], false, false);
      acc[7] = __builtin_amdgcn_wmma_f32_16x16x32_bf16(
          false, A0[1].v, false, B0[3].v, (short)0, acc[7], false, false);

      A0[0] = A1[0]; A0[1] = A1[1];
      B0[0] = B1[0]; B0[1] = B1[1]; B0[2] = B1[2]; B0[3] = B1[3];
    }

    // Scan the eight 16x16 subtiles (2 row bands x 4 col subtiles).
    // Lane owns row lr of each band; kh==0 scans cols 0..7, kh==1 cols 8..15.
#pragma unroll
    for (int t = 0; t < 2; ++t) {
      float b1 = t ? c_b1 : a_b1, b2 = t ? c_b2 : a_b2;
      int   i1 = t ? c_i1 : a_i1, i2 = t ? c_i2 : a_i2;
      const int grow = t ? grow1 : grow0;
#pragma unroll
      for (int s = 0; s < 4; ++s) {
#pragma unroll
        for (int e = 0; e < 8; ++e)
          tile[wave][e + 8 * kh][lr] = acc[t * 4 + s][e]; // VGPR e -> M=e+8*half, N=lane&15
        __builtin_amdgcn_wave_barrier();
#pragma unroll
        for (int n = 0; n < 8; ++n) {
          const int nn = kh * 8 + n;
          const int gcol = cb + s * 16 + nn;
          float v = tile[wave][lr][nn];
          v = (gcol == grow) ? -3.0f : v;         // exclude diagonal (ref: -1)
          const bool t1 = v > b1;
          const float c2v = t1 ? b1 : v;
          const int   c2i = t1 ? i1 : gcol;
          const bool t2 = c2v > b2;
          b2 = t2 ? c2v : b2;
          i2 = t2 ? c2i : i2;
          b1 = t1 ? v : b1;
          i1 = t1 ? gcol : i1;
        }
        __builtin_amdgcn_wave_barrier();
      }
      if (t) { c_b1 = b1; c_b2 = b2; c_i1 = i1; c_i2 = i2; }
      else   { a_b1 = b1; a_b2 = b2; a_i1 = i1; a_i2 = i2; }
    }
  }

  // Merge the two column-halves of each row (lane <-> lane^16), index
  // tie-break, then store partial top-2 for both row bands.
#pragma unroll
  for (int t = 0; t < 2; ++t) {
    float b1 = t ? c_b1 : a_b1, b2 = t ? c_b2 : a_b2;
    int   i1 = t ? c_i1 : a_i1, i2 = t ? c_i2 : a_i2;
    const float ob1 = __shfl_xor(b1, 16, 32);
    const int   oi1 = __shfl_xor(i1, 16, 32);
    const float ob2 = __shfl_xor(b2, 16, 32);
    const int   oi2 = __shfl_xor(i2, 16, 32);
    if (lane < 16) {
      if (better(ob1, oi1, b1, i1)) { b2 = b1; i2 = i1; b1 = ob1; i1 = oi1; }
      else if (better(ob1, oi1, b2, i2)) { b2 = ob1; i2 = oi1; }
      if (better(ob2, oi2, b2, i2)) { b2 = ob2; i2 = oi2; }  // ob2 only takes 2nd

      const int grow = t ? grow1 : grow0;
      const size_t base = ((size_t)grow * NCHUNKS + blockIdx.y) * 2;
      pVal[base]     = b1; pIdx[base]     = i1;
      pVal[base + 1] = b2; pIdx[base + 1] = i2;
    }
  }
}

// ---------------------------------------------------------------------------
// Kernel 3: merge per-chunk top-2 candidates into final top-2 indices
// ---------------------------------------------------------------------------
__global__ void __launch_bounds__(256)
merge_top2(const float* __restrict__ pVal, const int* __restrict__ pIdx,
           int* __restrict__ topIdx) {
  const int row = blockIdx.x * 256 + threadIdx.x;
  if (row >= B_ROWS) return;
  float b1 = -3.f, b2 = -3.f; int i1 = 0x7fffffff, i2 = 0x7fffffff;
  for (int c = 0; c < NCHUNKS; ++c) {
    const size_t base = ((size_t)row * NCHUNKS + c) * 2;
    for (int k = 0; k < 2; ++k) {
      const float v = pVal[base + k];
      const int idx = pIdx[base + k];
      if (better(v, idx, b1, i1)) { b2 = b1; i2 = i1; b1 = v; i1 = idx; }
      else if (better(v, idx, b2, i2)) { b2 = v; i2 = idx; }
    }
  }
  topIdx[row * 2]     = i1;
  topIdx[row * 2 + 1] = i2;
}

// ---------------------------------------------------------------------------
// Kernel 4: exact fp32 pair distance -> loss*gate, gate per (row, k)
// ---------------------------------------------------------------------------
__global__ void __launch_bounds__(256)
pair_loss(const float* __restrict__ x, const float* __restrict__ invn,
          const int* __restrict__ topIdx, float* __restrict__ lg,
          float* __restrict__ gt) {
  const int row = blockIdx.x;
  const float4* xi4 = reinterpret_cast<const float4*>(x + (size_t)row * DIMS);
  const float inv_i = invn[row];
  const float4 a4 = xi4[threadIdx.x];
  __shared__ float red[2][8];
  for (int k = 0; k < 2; ++k) {
    const int j = topIdx[row * 2 + k];
    const float4* xj4 = reinterpret_cast<const float4*>(x + (size_t)j * DIMS);
    const float inv_j = invn[j];
    const float4 c4 = xj4[threadIdx.x];
    float dx = a4.x * inv_i - c4.x * inv_j + EPSF;   // ref adds eps to the diff
    float dy = a4.y * inv_i - c4.y * inv_j + EPSF;
    float dz = a4.z * inv_i - c4.z * inv_j + EPSF;
    float dw = a4.w * inv_i - c4.w * inv_j + EPSF;
    float s = dx * dx + dy * dy + dz * dz + dw * dw;
    for (int o = 16; o > 0; o >>= 1) s += __shfl_down(s, o, 32);
    if ((threadIdx.x & 31) == 0) red[k][threadIdx.x >> 5] = s;
  }
  __syncthreads();
  if (threadIdx.x < 2) {
    float t = 0.f;
    for (int w = 0; w < 8; ++w) t += red[threadIdx.x][w];
    const float dist = sqrtf(t);
    const float loss = -logf(dist + EPSF);
    const float gate = 1.f / (1.f + expf(-(loss - GATE_THRESHOLD_F) / GATE_ALPHA_F));
    lg[row * 2 + threadIdx.x] = loss * gate;
    gt[row * 2 + threadIdx.x] = gate;
  }
}

// ---------------------------------------------------------------------------
// Kernel 5: deterministic single-block reduction -> scalar output
// ---------------------------------------------------------------------------
__global__ void __launch_bounds__(256)
final_reduce(const float* __restrict__ lg, const float* __restrict__ gt,
             float* __restrict__ out) {
  __shared__ float s1[256], s2[256];
  float a = 0.f, b = 0.f;
  for (int i = threadIdx.x; i < B_ROWS * 2; i += 256) { a += lg[i]; b += gt[i]; }
  s1[threadIdx.x] = a; s2[threadIdx.x] = b;
  __syncthreads();
  for (int o = 128; o > 0; o >>= 1) {
    if (threadIdx.x < o) {
      s1[threadIdx.x] += s1[threadIdx.x + o];
      s2[threadIdx.x] += s2[threadIdx.x + o];
    }
    __syncthreads();
  }
  if (threadIdx.x == 0) {
    const float wm = s1[0] / (float)(B_ROWS * 2);            // weighted_mean
    const float gm = s1[0] / fmaxf(s2[0], 1.f);              // gated_mean
    out[0] = 0.5f * wm + 0.5f * gm;
  }
}

// ---------------------------------------------------------------------------
extern "C" void kernel_launch(void* const* d_in, const int* in_sizes, int n_in,
                              void* d_out, int out_size, void* d_ws, size_t ws_size,
                              hipStream_t stream) {
  (void)in_sizes; (void)n_in; (void)out_size; (void)ws_size;
  const float* x = (const float*)d_in[0];
  float* out = (float*)d_out;

  char* ws = (char*)d_ws;
  size_t off = 0;
  auto alloc = [&](size_t bytes) -> void* {
    void* p = ws + off;
    off = (off + bytes + 255) & ~(size_t)255;
    return p;
  };
  float*          invn   = (float*)alloc((size_t)B_ROWS * sizeof(float));
  unsigned short* xnbf   = (unsigned short*)alloc((size_t)B_ROWS * DIMS * sizeof(unsigned short));
  float*          pVal   = (float*)alloc((size_t)B_ROWS * NCHUNKS * 2 * sizeof(float));
  int*            pIdx   = (int*)alloc((size_t)B_ROWS * NCHUNKS * 2 * sizeof(int));
  int*            topIdx = (int*)alloc((size_t)B_ROWS * 2 * sizeof(int));
  float*          lg     = (float*)alloc((size_t)B_ROWS * 2 * sizeof(float));
  float*          gt     = (float*)alloc((size_t)B_ROWS * 2 * sizeof(float));

  normalize_bf16<<<B_ROWS, 256, 0, stream>>>(x, xnbf, invn);

  dim3 grid2(B_ROWS / (ROWS_PER_WAVE * WAVES_PER_WG), NCHUNKS);   // (32, 16)
  koleo_topk_wmma<<<grid2, 256, 0, stream>>>(xnbf, pVal, pIdx);

  merge_top2<<<B_ROWS / 256, 256, 0, stream>>>(pVal, pIdx, topIdx);
  pair_loss<<<B_ROWS, 256, 0, stream>>>(x, invn, topIdx, lg, gt);
  final_reduce<<<1, 256, 0, stream>>>(lg, gt, out);
}
